// Albert_10917806866779
// MI455X (gfx1250) — matrix operand order repaired and checked by reading, not compiled, not run on hardware
//
#include <hip/hip_runtime.h>
#include <hip/hip_bf16.h>
#include <math.h>

typedef __attribute__((ext_vector_type(16))) _Float16 v16h;
typedef __attribute__((ext_vector_type(8)))  float    v8f;
typedef __attribute__((ext_vector_type(4)))  _Float16 v4h;

#define N_LAYERS 6
#define E_DIM 128
#define H_DIM 768
#define N_HEADS 12
#define D_FFN 3072
#define SEQ_L 512
#define N_WORD 30000
#define BATCH 4
#define D_HEAD 64
#define BL (BATCH*SEQ_L)   // 2048 token rows
#define LM1 (SEQ_L-1)      // 511

static __device__ __forceinline__ v8f wmma_f16(v16h a, v16h b, v8f c) {
    return __builtin_amdgcn_wmma_f32_16x16x32_f16(false, a, false, b, (short)0, c, false, false);
}

static __device__ __forceinline__ v4h cvt4(float4 v) {
    v4h h;
    h[0] = (_Float16)v.x; h[1] = (_Float16)v.y;
    h[2] = (_Float16)v.z; h[3] = (_Float16)v.w;
    return h;
}

// ---------------------------------------------------------------------------
// Embedding: x[t,:] = tok_emb[batch[t]] @ to_hid_w + to_hid_b + sinusoidal PE
// ---------------------------------------------------------------------------
__global__ __launch_bounds__(256)
void embed_kernel(const int* __restrict__ batch, const float* __restrict__ tok_emb,
                  const float* __restrict__ w, const float* __restrict__ bias,
                  float* __restrict__ x) {
    __shared__ float emb[E_DIM];
    const int t = blockIdx.x;           // token row 0..BL-1
    const int l = t % SEQ_L;
    const int tok = batch[t];
    for (int e = threadIdx.x; e < E_DIM; e += blockDim.x)
        emb[e] = tok_emb[(size_t)tok * E_DIM + e];
    __syncthreads();
    for (int j = threadIdx.x; j < H_DIM; j += blockDim.x) {
        float acc = bias[j];
        #pragma unroll 8
        for (int e = 0; e < E_DIM; ++e)
            acc += emb[e] * w[(size_t)e * H_DIM + j];
        const int pi = j >> 1;
        const float inv = expf(-((float)(2 * pi) / (float)H_DIM) * 9.210340371976184f);
        const float tv = (float)l * inv;
        acc += (j & 1) ? cosf(tv) : sinf(tv);
        x[(size_t)t * H_DIM + j] = acc;
    }
}

// ---------------------------------------------------------------------------
// Generic WMMA GEMM: C(MxN) = act(A(MxK) * B + bias)
//   BTRANS=0: B is KxN row-major; BTRANS=1: B is NxK row-major (B^T logical)
//   ACT: 0=none, 1=gelu (x*sigmoid(1.702x))
//   GUARD: bounds checks on M/N (only needed for the 2044x30000 logits GEMM)
// Tile 128x128, K-step 32, 256 threads (8 waves; each wave: 32x64 = 2x4 frags)
// Register-pipelined staging: next K-slice loaded to VGPRs during WMMA.
// ---------------------------------------------------------------------------
template <int BTRANS, int ACT, int GUARD>
__global__ __launch_bounds__(256)
void gemm_kernel(const float* __restrict__ A, const float* __restrict__ Bm,
                 const float* __restrict__ bias, float* __restrict__ C,
                 int M, int N, int K) {
    __shared__ alignas(16) _Float16 As[128 * 32];   // [m][k]
    __shared__ alignas(16) _Float16 Bs[128 * 32];   // [n][k]  (transposed staging)
    const int tid  = threadIdx.x;
    const int lane = tid & 31, wave = tid >> 5;
    const int wm = wave >> 1, wn = wave & 1;
    const int l15 = lane & 15, hi = lane >> 4;
    const int m0 = blockIdx.y * 128, n0 = blockIdx.x * 128;

    v8f acc[2][4];
    #pragma unroll
    for (int i = 0; i < 2; ++i)
        #pragma unroll
        for (int j = 0; j < 4; ++j)
            #pragma unroll
            for (int r = 0; r < 8; ++r) acc[i][j][r] = 0.0f;

    // staging geometry
    const int srow = tid >> 1;           // 0..127  (A rows / B^T rows)
    const int sk0  = (tid & 1) * 16;     // 0 or 16
    const int bk   = tid >> 5;           // 0..7    (B normal: k within pass)
    const int bn4  = (tid & 31) * 4;     // 0..124  (B normal: 4 cols)

    float4 aR[4], bR[4];
    const float4 f4z = make_float4(0.f, 0.f, 0.f, 0.f);

    // ---- register loads for K-slice kt ----
    auto noop = 0; (void)noop;
#define LOAD_A(kt)                                                              \
    {                                                                           \
        const int mg = m0 + srow;                                               \
        if (!GUARD || mg < M) {                                                 \
            const float* ap = A + (size_t)mg * K + (kt) + sk0;                  \
            _Pragma("unroll")                                                   \
            for (int e = 0; e < 4; ++e) aR[e] = *(const float4*)(ap + 4 * e);   \
        } else {                                                                \
            _Pragma("unroll")                                                   \
            for (int e = 0; e < 4; ++e) aR[e] = f4z;                            \
        }                                                                       \
    }
#define LOAD_B(kt)                                                              \
    if (BTRANS) {                                                               \
        const int ng = n0 + srow;                                               \
        if (!GUARD || ng < N) {                                                 \
            const float* bp = Bm + (size_t)ng * K + (kt) + sk0;                 \
            _Pragma("unroll")                                                   \
            for (int e = 0; e < 4; ++e) bR[e] = *(const float4*)(bp + 4 * e);   \
        } else {                                                                \
            _Pragma("unroll")                                                   \
            for (int e = 0; e < 4; ++e) bR[e] = f4z;                            \
        }                                                                       \
    } else {                                                                    \
        _Pragma("unroll")                                                       \
        for (int p = 0; p < 4; ++p) {                                           \
            const int kk = (kt) + p * 8 + bk;                                   \
            bR[p] = *(const float4*)(Bm + (size_t)kk * N + n0 + bn4);           \
        }                                                                       \
    }
#define STORE_AB()                                                              \
    {                                                                           \
        _Pragma("unroll")                                                       \
        for (int e = 0; e < 4; ++e)                                             \
            *(v4h*)&As[srow * 32 + sk0 + 4 * e] = cvt4(aR[e]);                  \
        if (BTRANS) {                                                           \
            _Pragma("unroll")                                                   \
            for (int e = 0; e < 4; ++e)                                         \
                *(v4h*)&Bs[srow * 32 + sk0 + 4 * e] = cvt4(bR[e]);              \
        } else {                                                                \
            _Pragma("unroll")                                                   \
            for (int p = 0; p < 4; ++p) {                                       \
                const int kl = p * 8 + bk;                                      \
                Bs[(bn4 + 0) * 32 + kl] = (_Float16)bR[p].x;                    \
                Bs[(bn4 + 1) * 32 + kl] = (_Float16)bR[p].y;                    \
                Bs[(bn4 + 2) * 32 + kl] = (_Float16)bR[p].z;                    \
                Bs[(bn4 + 3) * 32 + kl] = (_Float16)bR[p].w;                    \
            }                                                                   \
        }                                                                       \
    }

    LOAD_A(0)
    LOAD_B(0)
    STORE_AB()

    for (int kt = 0; kt < K; kt += 32) {
        __syncthreads();
        const bool more = (kt + 32 < K);
        if (more) { LOAD_A(kt + 32) LOAD_B(kt + 32) }
        // ---- fragments + WMMA from LDS ----
        v16h af[2];
        #pragma unroll
        for (int fm = 0; fm < 2; ++fm) {
            const _Float16* ap = &As[(wm * 32 + fm * 16 + l15) * 32 + hi * 16];
            #pragma unroll
            for (int e = 0; e < 8; ++e) { af[fm][e] = ap[e]; af[fm][8 + e] = ap[8 + e]; }
        }
        #pragma unroll
        for (int fn = 0; fn < 4; ++fn) {
            v16h bf;
            const _Float16* bp = &Bs[(wn * 64 + fn * 16 + l15) * 32 + hi * 16];
            #pragma unroll
            for (int e = 0; e < 16; ++e) bf[e] = bp[e];
            #pragma unroll
            for (int fm = 0; fm < 2; ++fm)
                acc[fm][fn] = wmma_f16(af[fm], bf, acc[fm][fn]);
        }
        __syncthreads();
        if (more) { STORE_AB() }
    }
#undef LOAD_A
#undef LOAD_B
#undef STORE_AB

    // ---- epilogue: bias + activation + store ----
    #pragma unroll
    for (int fn = 0; fn < 4; ++fn) {
        const int col = n0 + wn * 64 + fn * 16 + l15;
        if (GUARD && col >= N) continue;
        const float bv = bias ? bias[col] : 0.0f;
        #pragma unroll
        for (int fm = 0; fm < 2; ++fm) {
            #pragma unroll
            for (int r = 0; r < 8; ++r) {
                const int row = m0 + wm * 32 + fm * 16 + hi * 8 + r;
                if (!GUARD || row < M) {
                    float v = acc[fm][fn][r] + bv;
                    if (ACT == 1) v = v / (1.0f + expf(-1.702f * v));  // gelu approx
                    C[(size_t)row * N + col] = v;
                }
            }
        }
    }
}

// ---------------------------------------------------------------------------
// Residual + LayerNorm (row = 768). resid may be null. In-place safe per row.
// ---------------------------------------------------------------------------
__global__ __launch_bounds__(256)
void ln_kernel(const float* __restrict__ src, const float* __restrict__ resid,
               const float* __restrict__ g, const float* __restrict__ bta,
               float* __restrict__ dst) {
    __shared__ float red[256];
    const int row = blockIdx.x, tid = threadIdx.x;
    float v[3];
    float s = 0.0f;
    #pragma unroll
    for (int c = 0; c < 3; ++c) {
        const int idx = tid + c * 256;
        float t = src[(size_t)row * H_DIM + idx];
        if (resid) t += resid[(size_t)row * H_DIM + idx];
        v[c] = t; s += t;
    }
    red[tid] = s; __syncthreads();
    for (int st = 128; st > 0; st >>= 1) { if (tid < st) red[tid] += red[tid + st]; __syncthreads(); }
    const float mean = red[0] / (float)H_DIM;
    __syncthreads();
    float s2 = 0.0f;
    #pragma unroll
    for (int c = 0; c < 3; ++c) { const float d = v[c] - mean; s2 += d * d; }
    red[tid] = s2; __syncthreads();
    for (int st = 128; st > 0; st >>= 1) { if (tid < st) red[tid] += red[tid + st]; __syncthreads(); }
    const float rstd = rsqrtf(red[0] / (float)H_DIM + 1e-5f);
    #pragma unroll
    for (int c = 0; c < 3; ++c) {
        const int idx = tid + c * 256;
        dst[(size_t)row * H_DIM + idx] = (v[c] - mean) * rstd * g[idx] + bta[idx];
    }
}

// kc[b,h,j] = cb_h . k[b,h,j,:]
__global__ void kc_kernel(const float* __restrict__ kb, const float* __restrict__ cb_l,
                          float* __restrict__ kc) {
    const int idx = blockIdx.x * blockDim.x + threadIdx.x;
    if (idx >= BATCH * N_HEADS * SEQ_L) return;
    const int j = idx & (SEQ_L - 1);
    const int h = (idx >> 9) % N_HEADS;
    const int b = idx / (N_HEADS * SEQ_L);
    const float* kp = kb + (size_t)(b * SEQ_L + j) * H_DIM + h * D_HEAD;
    const float* cp = cb_l + h * D_HEAD;
    float acc = 0.0f;
    #pragma unroll
    for (int d = 0; d < D_HEAD; ++d) acc += cp[d] * kp[d];
    kc[idx] = acc;
}

// dr[h,jr] = pb_h . Qp[h,jr,:]
__global__ void dr_kernel(const float* __restrict__ qp, const float* __restrict__ pb_l,
                          float* __restrict__ dr) {
    const int idx = blockIdx.x * blockDim.x + threadIdx.x;
    if (idx >= N_HEADS * SEQ_L) return;
    const int jr = idx & (SEQ_L - 1);
    const int h  = idx >> 9;
    const float* qr = qp + (size_t)jr * H_DIM + h * D_HEAD;
    const float* pp = pb_l + h * D_HEAD;
    float acc = 0.0f;
    #pragma unroll
    for (int d = 0; d < D_HEAD; ++d) acc += pp[d] * qr[d];
    dr[idx] = acc;
}

// ---------------------------------------------------------------------------
// Flash-style rel-pos attention. One block per (b, h, 64-row i-tile).
// A[i,j] = q_i.k_j + cb.k_j + [j<=i]( q_i.Qp[511-i+j] + Dr[511-i+j] )
// Online softmax over all j (no mask on content terms), O = softmax(A) V.
// ---------------------------------------------------------------------------
__global__ __launch_bounds__(256)
void attn_kernel(const float* __restrict__ qb, const float* __restrict__ kb,
                 const float* __restrict__ vb, const float* __restrict__ qp,
                 const float* __restrict__ kc, const float* __restrict__ dr,
                 float* __restrict__ ob) {
    const int blk = blockIdx.x;
    const int it = blk & 7;
    const int h  = (blk >> 3) % N_HEADS;
    const int b  = blk / (8 * N_HEADS);
    const int i0 = it * 64;

    __shared__ alignas(16) _Float16 qs[64 * 64];     // [i][d]
    __shared__ alignas(16) _Float16 kts[64 * 64];    // [j][d]  (= B^T staged for S)
    __shared__ alignas(16) _Float16 vts[64 * 64];    // [d][j]  (= B^T staged for PV)
    __shared__ alignas(16) _Float16 ps[64 * 64];     // P tile f16 [i][j]
    __shared__ float ss[64 * 64];                    // score tile f32
    __shared__ float mrow[64], lrow[64], arow[64];

    const int tid = threadIdx.x, lane = tid & 31, wave = tid >> 5;
    const int wm = wave >> 1, wn = wave & 1;
    const int l15 = lane & 15, hi = lane >> 4;
    const int r4 = tid >> 2, c0 = (tid & 3) * 16;    // staging coords

    // load Q tile (float4 -> packed f16)
    {
        const float* qptr = qb + (size_t)(b * SEQ_L + i0 + r4) * H_DIM + h * D_HEAD + c0;
        #pragma unroll
        for (int e = 0; e < 4; ++e)
            *(v4h*)&qs[r4 * 64 + c0 + 4 * e] = cvt4(*(const float4*)(qptr + 4 * e));
    }
    if (tid < 64) { mrow[tid] = -3.0e38f; lrow[tid] = 0.0f; }
    v8f oacc[2];
    #pragma unroll
    for (int fn = 0; fn < 2; ++fn)
        #pragma unroll
        for (int r = 0; r < 8; ++r) oacc[fn][r] = 0.0f;
    __syncthreads();

    for (int jt = 0; jt < 8; ++jt) {
        const int j0 = jt * 64;
        // load K (vectorized) and V (transposed scatter) tiles
        {
            const float* kp = kb + (size_t)(b * SEQ_L + j0 + r4) * H_DIM + h * D_HEAD + c0;
            const float* vp = vb + (size_t)(b * SEQ_L + j0 + r4) * H_DIM + h * D_HEAD + c0;
            #pragma unroll
            for (int e = 0; e < 4; ++e) {
                *(v4h*)&kts[r4 * 64 + c0 + 4 * e] = cvt4(*(const float4*)(kp + 4 * e));
                const float4 vv = *(const float4*)(vp + 4 * e);
                vts[(c0 + 4 * e + 0) * 64 + r4] = (_Float16)vv.x;
                vts[(c0 + 4 * e + 1) * 64 + r4] = (_Float16)vv.y;
                vts[(c0 + 4 * e + 2) * 64 + r4] = (_Float16)vv.z;
                vts[(c0 + 4 * e + 3) * 64 + r4] = (_Float16)vv.w;
            }
        }
        __syncthreads();
        // S = Q K^T  (per wave: rows wm*16.., cols wn*32 + fn*16)
        v8f sf[2];
        #pragma unroll
        for (int fn = 0; fn < 2; ++fn)
            #pragma unroll
            for (int r = 0; r < 8; ++r) sf[fn][r] = 0.0f;
        #pragma unroll
        for (int ks = 0; ks < 64; ks += 32) {
            v16h aq;
            const _Float16* ap = &qs[(wm * 16 + l15) * 64 + ks + hi * 16];
            #pragma unroll
            for (int e = 0; e < 8; ++e) { aq[e] = ap[e]; aq[8 + e] = ap[8 + e]; }
            #pragma unroll
            for (int fn = 0; fn < 2; ++fn) {
                v16h bk;
                const _Float16* bp = &kts[(wn * 32 + fn * 16 + l15) * 64 + ks + hi * 16];
                #pragma unroll
                for (int e = 0; e < 16; ++e) bk[e] = bp[e];
                sf[fn] = wmma_f16(aq, bk, sf[fn]);
            }
        }
        #pragma unroll
        for (int fn = 0; fn < 2; ++fn)
            #pragma unroll
            for (int r = 0; r < 8; ++r)
                ss[(wm * 16 + hi * 8 + r) * 64 + wn * 32 + fn * 16 + l15] = sf[fn][r];
        __syncthreads();
        // bias epilogue: content bias + shifted position terms
        {
            const int ii = r4, jj0 = c0;
            const int ig = i0 + ii;
            float qrow[D_HEAD];
            #pragma unroll
            for (int d = 0; d < D_HEAD; ++d) qrow[d] = (float)qs[ii * 64 + d];
            #pragma unroll
            for (int e = 0; e < 16; ++e) {
                const int jj = jj0 + e, jg = j0 + jj;
                float s = ss[ii * 64 + jj] + kc[(size_t)(b * N_HEADS + h) * SEQ_L + jg];
                if (jg <= ig) {
                    const int jr = (SEQ_L - 1) - ig + jg;
                    const float* qpr = qp + (size_t)jr * H_DIM + h * D_HEAD;
                    float acc = dr[h * SEQ_L + jr];
                    #pragma unroll
                    for (int d = 0; d < D_HEAD; ++d) acc += qrow[d] * qpr[d];
                    s += acc;
                }
                ss[ii * 64 + jj] = s;
            }
        }
        __syncthreads();
        // online softmax row stats
        if (tid < 64) {
            const float mo = mrow[tid];
            float mt = -3.0e38f;
            for (int j = 0; j < 64; ++j) mt = fmaxf(mt, ss[tid * 64 + j]);
            const float mn = fmaxf(mo, mt);
            const float al = expf(mo - mn);
            float ls = 0.0f;
            for (int j = 0; j < 64; ++j) {
                const float p = expf(ss[tid * 64 + j] - mn);
                ps[tid * 64 + j] = (_Float16)p;
                ls += p;
            }
            lrow[tid] = lrow[tid] * al + ls;
            mrow[tid] = mn; arow[tid] = al;
        }
        __syncthreads();
        // rescale O accumulators
        #pragma unroll
        for (int fn = 0; fn < 2; ++fn)
            #pragma unroll
            for (int r = 0; r < 8; ++r) oacc[fn][r] *= arow[wm * 16 + hi * 8 + r];
        // O += P V
        #pragma unroll
        for (int ks = 0; ks < 64; ks += 32) {
            v16h apf;
            const _Float16* ap = &ps[(wm * 16 + l15) * 64 + ks + hi * 16];
            #pragma unroll
            for (int e = 0; e < 8; ++e) { apf[e] = ap[e]; apf[8 + e] = ap[8 + e]; }
            #pragma unroll
            for (int fn = 0; fn < 2; ++fn) {
                v16h bv;
                const _Float16* bp = &vts[(wn * 32 + fn * 16 + l15) * 64 + ks + hi * 16];
                #pragma unroll
                for (int e = 0; e < 16; ++e) bv[e] = bp[e];
                oacc[fn] = wmma_f16(apf, bv, oacc[fn]);
            }
        }
        __syncthreads();
    }
    // store O / l
    #pragma unroll
    for (int fn = 0; fn < 2; ++fn)
        #pragma unroll
        for (int r = 0; r < 8; ++r) {
            const int irow = wm * 16 + hi * 8 + r;
            const int dcol = wn * 32 + fn * 16 + l15;
            ob[(size_t)(b * SEQ_L + i0 + irow) * H_DIM + h * D_HEAD + dcol] =
                oacc[fn][r] / lrow[irow];
        }
}

// SOP head: C[b, 0:2] from token-0 rows
__global__ void sop_kernel(const float* __restrict__ x, const float* __restrict__ w,
                           const float* __restrict__ bias, float* __restrict__ out) {
    const int t = threadIdx.x;
    if (t >= BATCH * 2) return;
    const int b = t >> 1, c = t & 1;
    const float* xr = x + (size_t)(b * SEQ_L) * H_DIM;
    float acc = bias[c];
    for (int hh = 0; hh < H_DIM; ++hh) acc += xr[hh] * w[hh * 2 + c];
    out[t] = acc;
}

// drop token 0 of each batch: epack[b*511 + (l-1), :] = e[b*512 + l, :]
__global__ void pack_kernel(const float* __restrict__ ebuf, float* __restrict__ epk) {
    const int idx = blockIdx.x * blockDim.x + threadIdx.x;
    if (idx >= BATCH * LM1 * E_DIM) return;
    const int col = idx % E_DIM;
    const int row = idx / E_DIM;
    const int b = row / LM1;
    const int l = row % LM1 + 1;
    epk[idx] = ebuf[(size_t)(b * SEQ_L + l) * E_DIM + col];
}

// ---------------------------------------------------------------------------
extern "C" void kernel_launch(void* const* d_in, const int* in_sizes, int n_in,
                              void* d_out, int out_size, void* d_ws, size_t ws_size,
                              hipStream_t stream) {
    const int*   batch    = (const int*)  d_in[0];
    const float* tok_emb  = (const float*)d_in[2];
    const float* to_hid_w = (const float*)d_in[3];
    const float* to_hid_b = (const float*)d_in[4];
    const float* R        = (const float*)d_in[5];
    const float* Wq       = (const float*)d_in[6];
    const float* Wke      = (const float*)d_in[7];
    const float* Wv       = (const float*)d_in[8];
    const float* Wkr      = (const float*)d_in[9];
    const float* cb       = (const float*)d_in[10];
    const float* pb       = (const float*)d_in[11];
    const float* Wo_w     = (const float*)d_in[12];
    const float* Wo_b     = (const float*)d_in[13];
    const float* ln1_g    = (const float*)d_in[14];
    const float* ln1_b    = (const float*)d_in[15];
    const float* ff1_w    = (const float*)d_in[16];
    const float* ff1_b    = (const float*)d_in[17];
    const float* ff2_w    = (const float*)d_in[18];
    const float* ff2_b    = (const float*)d_in[19];
    const float* ln2_g    = (const float*)d_in[20];
    const float* ln2_b    = (const float*)d_in[21];
    const float* osp_w    = (const float*)d_in[22];
    const float* osp_b    = (const float*)d_in[23];
    const float* mlm_w    = (const float*)d_in[24];
    const float* mlm_b    = (const float*)d_in[25];
    const float* mlm_ln_g = (const float*)d_in[26];
    const float* mlm_ln_b = (const float*)d_in[27];
    const float* to_emb_w = (const float*)d_in[28];
    const float* to_emb_b = (const float*)d_in[29];
    const float* out_b    = (const float*)d_in[30];
    float* out = (float*)d_out;

    // workspace partition (fp32)
    float* ws = (float*)d_ws;
    size_t o = 0;
    float* xb    = ws + o; o += (size_t)BL * H_DIM;
    float* qbuf  = ws + o; o += (size_t)BL * H_DIM;
    float* kbuf  = ws + o; o += (size_t)BL * H_DIM;
    float* vbuf  = ws + o; o += (size_t)BL * H_DIM;
    float* obuf  = ws + o; o += (size_t)BL * H_DIM;
    float* tbuf  = ws + o; o += (size_t)BL * H_DIM;
    float* fbuf  = ws + o; o += (size_t)BL * D_FFN;
    float* qpbuf = ws + o; o += (size_t)SEQ_L * H_DIM;
    float* kcb   = ws + o; o += (size_t)BATCH * N_HEADS * SEQ_L;
    float* drb   = ws + o; o += (size_t)N_HEADS * SEQ_L;
    float* ebuf  = ws + o; o += (size_t)BL * E_DIM;
    float* epk   = ws + o; o += (size_t)BATCH * LM1 * E_DIM;
    (void)ws_size; (void)in_sizes; (void)n_in; (void)out_size;

    // dispatch helpers (3 instantiations: plain / gelu / guarded-B^T)
    auto gemm = [&](const float* A, const float* Bm, const float* bias, float* C,
                    int M, int N, int K) {
        dim3 grid((N + 127) / 128, (M + 127) / 128);
        gemm_kernel<0, 0, 0><<<grid, 256, 0, stream>>>(A, Bm, bias, C, M, N, K);
    };
    auto gemm_gelu = [&](const float* A, const float* Bm, const float* bias, float* C,
                         int M, int N, int K) {
        dim3 grid((N + 127) / 128, (M + 127) / 128);
        gemm_kernel<0, 1, 0><<<grid, 256, 0, stream>>>(A, Bm, bias, C, M, N, K);
    };
    auto gemm_bt = [&](const float* A, const float* Bm, const float* bias, float* C,
                       int M, int N, int K) {
        dim3 grid((N + 127) / 128, (M + 127) / 128);
        gemm_kernel<1, 0, 1><<<grid, 256, 0, stream>>>(A, Bm, bias, C, M, N, K);
    };

    // embedding + positional encoding
    embed_kernel<<<BL, 256, 0, stream>>>(batch, tok_emb, to_hid_w, to_hid_b, xb);

    for (int i = 0; i < N_LAYERS; ++i) {
        const size_t HH = (size_t)H_DIM * H_DIM;
        gemm(xb, Wq  + i * HH, nullptr, qbuf, BL, H_DIM, H_DIM);
        gemm(xb, Wke + i * HH, nullptr, kbuf, BL, H_DIM, H_DIM);
        gemm(xb, Wv  + i * HH, nullptr, vbuf, BL, H_DIM, H_DIM);
        gemm(R + (size_t)i * SEQ_L * H_DIM, Wkr + i * HH, nullptr, qpbuf,
             SEQ_L, H_DIM, H_DIM);
        kc_kernel<<<(BATCH * N_HEADS * SEQ_L + 255) / 256, 256, 0, stream>>>(
            kbuf, cb + (size_t)i * H_DIM, kcb);
        dr_kernel<<<(N_HEADS * SEQ_L + 255) / 256, 256, 0, stream>>>(
            qpbuf, pb + (size_t)i * H_DIM, drb);
        attn_kernel<<<BATCH * N_HEADS * (SEQ_L / 64), 256, 0, stream>>>(
            qbuf, kbuf, vbuf, qpbuf, kcb, drb, obuf);
        gemm(obuf, Wo_w + i * HH, Wo_b + (size_t)i * H_DIM, tbuf, BL, H_DIM, H_DIM);
        ln_kernel<<<BL, 256, 0, stream>>>(xb, tbuf,
            ln1_g + (size_t)i * H_DIM, ln1_b + (size_t)i * H_DIM, xb);
        gemm_gelu(xb, ff1_w + (size_t)i * H_DIM * D_FFN, ff1_b + (size_t)i * D_FFN,
                  fbuf, BL, D_FFN, H_DIM);
        gemm(fbuf, ff2_w + (size_t)i * D_FFN * H_DIM, ff2_b + (size_t)i * H_DIM,
             tbuf, BL, H_DIM, D_FFN);
        ln_kernel<<<BL, 256, 0, stream>>>(xb, tbuf,
            ln2_g + (size_t)i * H_DIM, ln2_b + (size_t)i * H_DIM, xb);
    }

    // SOP head -> out[0:8]
    sop_kernel<<<1, 64, 0, stream>>>(xb, osp_w, osp_b, out);

    // MLM head: m = LN(gelu(x @ mlm_w + b)); e = m @ to_emb_w + b
    gemm_gelu(xb, mlm_w, mlm_b, tbuf, BL, H_DIM, H_DIM);
    ln_kernel<<<BL, 256, 0, stream>>>(tbuf, nullptr, mlm_ln_g, mlm_ln_b, obuf);
    gemm(obuf, to_emb_w, to_emb_b, ebuf, BL, E_DIM, H_DIM);
    pack_kernel<<<(BATCH * LM1 * E_DIM + 255) / 256, 256, 0, stream>>>(ebuf, epk);
    // tied logits: (2044 x 128) @ tok_emb^T (128 x 30000) + out_b -> out[8:]
    gemm_bt(epk, tok_emb, out_b, out + BATCH * 2, BATCH * LM1, N_WORD, E_DIM);
}
